// disciminativeAno_47261820125556
// MI455X (gfx1250) — matrix operand order repaired and checked by reading, not compiled, not run on hardware
//
#include <hip/hip_runtime.h>
#include <hip/hip_bf16.h>
#include <stdint.h>
#include <stddef.h>

// ---------------------------------------------------------------------------
// CDNA5 (gfx1250) VAE-anomaly pipeline.
// Implicit-GEMM templated on geometry + element types. f16 activations in HBM
// (halves the bandwidth-bound traffic), f16 WMMA with f32 accumulation:
// v_wmma_f32_16x16x32_f16, wave32, 8 waves/block, 128x32 block tile,
// 2 WMMAs per wave per K-step.
// ---------------------------------------------------------------------------

typedef __attribute__((ext_vector_type(16))) _Float16 v16h;
typedef __attribute__((ext_vector_type(4)))  _Float16 h4;
typedef __attribute__((ext_vector_type(8)))  float    v8f;

#define TMB 128   // block M tile
#define TNB 32    // block N tile
#define TK  32    // K step
#define LDA 36    // 8-byte aligned rows for vectorized staging
#define LDB 36

// MODE: 0 +bias | 1 relu(+bias) | 2 relu(+bias)*scale+off | 3 sigmoid(+bias)
template <typename TI, typename TO, int IH, int IW, int CI, int OH, int OW,
          int CO, int KH, int KW, int STRIDE, int PAD, int UPS, int FLIP, int MODE>
__global__ __launch_bounds__(256) void wmma_gemm(
    const TI* __restrict__ in, const float* __restrict__ w,
    const float* __restrict__ bias, const float* __restrict__ scale,
    const float* __restrict__ off, TO* __restrict__ out) {
  __shared__ _Float16 sA[TMB * LDA];  // 128 x 32 f16 A tile
  __shared__ _Float16 sB[TK * LDB];   // 32 x 32 f16 weight tile

  constexpr int  K    = KH * KW * CI;
  constexpr int  spat = OH * OW;
  constexpr int  kwci = KW * CI;
  constexpr int  nK   = (K + TK - 1) / TK;
  constexpr bool DENSE = (KH == 1) && (KW == 1) && (PAD == 0) &&
                         (UPS == 1) && (STRIDE == 1);   // A is contiguous

  const int tid  = threadIdx.x;
  const int wave = tid >> 5;
  const int lane = tid & 31;
  const int hi   = lane >> 4;
  const int ml   = lane & 15;

  const int mBlk = blockIdx.x * TMB;
  const int nBlk = blockIdx.y * TNB;
  const int mSub = (wave >> 1) * 16;  // 0..48, second tile at +64
  const int nSub = (wave & 1) * 16;

  v8f acc0 = {};
  v8f acc1 = {};

  for (int k0 = 0; k0 < nK; ++k0) {
    // ---- stage A tile ----
    if constexpr (DENSE) {
      // contiguous rows: vectorized 4-half copies (K is a multiple of 32 here)
      #pragma unroll
      for (int i = 0; i < (TMB * TK) / (256 * 4); ++i) {
        int idx = tid + i * 256;          // vec4 index
        int rl  = idx >> 3;               // 8 vec4 per row
        int kv  = (idx & 7) * 4;
        const TI* src = in + (size_t)(mBlk + rl) * CI + k0 * TK + kv;
        h4 v;
        #pragma unroll
        for (int c = 0; c < 4; ++c) v[c] = (_Float16)src[c];
        *(h4*)(&sA[rl * LDA + kv]) = v;
      }
    } else {
      // implicit im2col gather; batch the loads, then convert+store
      float vals[(TMB * TK) / 256];
      #pragma unroll
      for (int i = 0; i < (TMB * TK) / 256; ++i) {
        int idx = tid + i * 256;
        int rl  = idx >> 5;               // 0..127
        int kk  = idx & 31;
        int k   = k0 * TK + kk;
        vals[i] = 0.f;
        if (K % TK == 0 || k < K) {
          int row = mBlk + rl;
          int b   = row / spat;           // constexpr divisors -> shifts/mults
          int rr  = row - b * spat;
          int oy  = rr / OW;
          int ox  = rr - oy * OW;
          int ky  = k / kwci;
          int t   = k - ky * kwci;
          int kx  = t / CI;
          int ci  = t - kx * CI;
          int iyu = oy * STRIDE + ky - PAD;  // coordinate on upsampled grid
          int ixu = ox * STRIDE + kx - PAD;
          bool ok = (iyu >= 0) & (ixu >= 0) &
                    (iyu <= (IH - 1) * UPS) & (ixu <= (IW - 1) * UPS);
          if (UPS > 1) ok = ok & ((iyu % UPS) == 0) & ((ixu % UPS) == 0);
          if (ok) {
            int iy = iyu / UPS, ix = ixu / UPS;
            vals[i] = (float)in[(((size_t)b * IH + iy) * IW + ix) * CI + ci];
          }
        }
      }
      #pragma unroll
      for (int i = 0; i < (TMB * TK) / 256; ++i) {
        int idx = tid + i * 256;
        sA[(idx >> 5) * LDA + (idx & 31)] = (_Float16)vals[i];
      }
    }
    // ---- stage B tile (weights, spatial flip for convT) ----
    {
      float vals[(TK * TNB) / 256];
      #pragma unroll
      for (int i = 0; i < (TK * TNB) / 256; ++i) {
        int idx = tid + i * 256;
        int kk  = idx >> 5;
        int nl  = idx & 31;
        int k   = k0 * TK + kk;
        int n   = nBlk + nl;
        vals[i] = 0.f;
        if ((K % TK == 0 || k < K) && (CO % TNB == 0 || n < CO)) {
          int ky = k / kwci;
          int t  = k - ky * kwci;
          int kx = t / CI;
          int ci = t - kx * CI;
          if (FLIP) { ky = KH - 1 - ky; kx = KW - 1 - kx; }
          vals[i] = w[(((size_t)ky * KW + kx) * CI + ci) * CO + n];
        }
      }
      #pragma unroll
      for (int i = 0; i < (TK * TNB) / 256; ++i) {
        int idx = tid + i * 256;
        sB[(idx >> 5) * LDB + (idx & 31)] = (_Float16)vals[i];
      }
    }
    __syncthreads();

    // ---- WMMA fragments per CDNA5 16-bit layouts ----
    v16h a0, a1, b;
    #pragma unroll
    for (int e = 0; e < 16; ++e) {
      // A 16x32: lanes<16 hold K {0..7,16..23}, lanes>=16 hold K {8..15,24..31}
      int kA = e + ((e >= 8) ? 8 : 0) + hi * 8;
      a0[e] = sA[(mSub + ml) * LDA + kA];
      a1[e] = sA[(mSub + 64 + ml) * LDA + kA];
    }
    const int kRow = ml + hi * 16;   // B 32x16: lane indexes the N column, K in A's order
    #pragma unroll
    for (int e = 0; e < 16; ++e)
      b[e] = sB[(e + ((e >= 8) ? 8 : 0) + hi * 8) * LDB + nSub + ml];

    acc0 = __builtin_amdgcn_wmma_f32_16x16x32_f16(false, a0, false, b,
                                                  (short)0, acc0, false, false);
    acc1 = __builtin_amdgcn_wmma_f32_16x16x32_f16(false, a1, false, b,
                                                  (short)0, acc1, false, false);
    __syncthreads();
  }

  // ---- fused epilogue ----
  const int n = nBlk + nSub + ml;
  if (CO % TNB == 0 || n < CO) {
    const float bia = bias[n];
    const float sc  = (MODE == 2) ? scale[n] : 1.f;
    const float of  = (MODE == 2) ? off[n]   : 0.f;
    #pragma unroll
    for (int r = 0; r < 8; ++r) {
      // C/D layout: VGPR r -> M = r (lanes 0-15) / r+8 (lanes 16-31)
      int m0 = mBlk + mSub + r + 8 * hi;
      float v0 = acc0[r] + bia;
      float v1 = acc1[r] + bia;
      if (MODE == 1 || MODE == 2) { v0 = fmaxf(v0, 0.f); v1 = fmaxf(v1, 0.f); }
      if (MODE == 2) { v0 = v0 * sc + of; v1 = v1 * sc + of; }
      if (MODE == 3) {
        v0 = 1.f / (1.f + __expf(-v0));
        v1 = 1.f / (1.f + __expf(-v1));
      }
      out[(size_t)m0 * CO + n] = (TO)v0;
      out[(size_t)(m0 + 64) * CO + n] = (TO)v1;
    }
  }
}

template <typename TI, typename TO, int IH, int IW, int CI, int OH, int OW,
          int CO, int KH, int KW, int STRIDE, int PAD, int UPS, int FLIP, int MODE>
static void launch_gemm(const TI* in, const float* w, const float* bias,
                        const float* scale, const float* off, TO* out,
                        int NB, hipStream_t stream) {
  int M = NB * OH * OW;
  dim3 grid((unsigned)(M / TMB), (unsigned)((CO + TNB - 1) / TNB));
  hipLaunchKernelGGL(
      (wmma_gemm<TI, TO, IH, IW, CI, OH, OW, CO, KH, KW, STRIDE, PAD, UPS, FLIP, MODE>),
      grid, dim3(256), 0, stream, in, w, bias, scale, off, out);
}

// ---- reparameterization: z = zm + exp(0.5*zl)*eps (f16 in/out, f32 math) ----
__global__ void sampling_kernel(const _Float16* __restrict__ zm,
                                const _Float16* __restrict__ zl,
                                const float* __restrict__ eps,
                                _Float16* __restrict__ z, int n) {
  int i = blockIdx.x * blockDim.x + threadIdx.x;
  if (i < n)
    z[i] = (_Float16)((float)zm[i] + __expf(0.5f * (float)zl[i]) * eps[i]);
}

// ---- per-pixel MSE over channels: err[b,y,x] = mean_c (x - recon)^2 ----
__global__ void error_kernel(const float* __restrict__ x,
                             const _Float16* __restrict__ rec,
                             float* __restrict__ err, int n) {
  int i = blockIdx.x * blockDim.x + threadIdx.x;
  if (i < n) {
    float s = 0.f;
    #pragma unroll
    for (int c = 0; c < 3; ++c) {
      float d = x[(size_t)i * 3 + c] - (float)rec[(size_t)i * 3 + c];
      s += d * d;
    }
    err[i] = s * (1.f / 3.f);
  }
}

// ---- per-image Otsu scan over 81 thresholds (faithful to the TF source) ----
__global__ __launch_bounds__(256) void otsu_kernel(const float* __restrict__ err,
                                                   float* __restrict__ thresh) {
  __shared__ float red[8][4];
  const int b   = blockIdx.x;
  const int tid = threadIdx.x;
  const float4 ev = ((const float4*)(err + (size_t)b * 1024))[tid];
  float e[4] = {ev.x, ev.y, ev.z, ev.w};

  float best = 0.f, opti = 0.f;
  for (int it = 0; it < 81; ++it) {
    const float t = 0.1f + 0.01f * (float)it;
    float c0 = 0.f, c1 = 0.f, s0 = 0.f, s1 = 0.f;
    #pragma unroll
    for (int j = 0; j < 4; ++j) {
      if (e[j] < t) { s0 += e[j]; if (e[j] != 0.f) c0 += 1.f; }
      else          { s1 += e[j]; if (e[j] != 0.f) c1 += 1.f; }
    }
    for (int o = 16; o > 0; o >>= 1) {   // wave32 reduction
      c0 += __shfl_xor(c0, o);
      c1 += __shfl_xor(c1, o);
      s0 += __shfl_xor(s0, o);
      s1 += __shfl_xor(s1, o);
    }
    if ((tid & 31) == 0) {
      int wv = tid >> 5;
      red[wv][0] = c0; red[wv][1] = c1; red[wv][2] = s0; red[wv][3] = s1;
    }
    __syncthreads();
    if (tid == 0) {
      float C0 = 0.f, C1 = 0.f, S0 = 0.f, S1 = 0.f;
      #pragma unroll
      for (int wv = 0; wv < 8; ++wv) {
        C0 += red[wv][0]; C1 += red[wv][1]; S0 += red[wv][2]; S1 += red[wv][3];
      }
      float tot = C0 + C1;
      if (tot > 0.f) {                   // NaN guard mirrors jnp.where(isnan,...)
        float m0 = (C0 > 0.f) ? (S0 * (1.f / 1024.f)) * (tot / C0) : 0.f;
        float m1 = (C1 > 0.f) ? (S1 * (1.f / 1024.f)) * (tot / C1) : 0.f;
        float p0 = C0 / tot, p1 = C1 / tot;
        float sb = p0 * p1 * (m0 - m1) * (m0 - m1);
        if (sb >= best) { best = sb; opti = t; }  // >= : later thresholds win ties
      }
    }
    __syncthreads();
  }
  if (tid == 0) thresh[b] = opti;
}

// ---- final int32 mask: error < per-image threshold ----
__global__ void mask_kernel(const float* __restrict__ err, const float* __restrict__ thresh,
                            int* __restrict__ out, int n) {
  int i = blockIdx.x * blockDim.x + threadIdx.x;
  if (i < n) out[i] = (err[i] < thresh[i >> 10]) ? 1 : 0;
}

// ---------------------------------------------------------------------------

extern "C" void kernel_launch(void* const* d_in, const int* in_sizes, int n_in,
                              void* d_out, int out_size, void* d_ws, size_t ws_size,
                              hipStream_t stream) {
  (void)in_sizes; (void)n_in; (void)out_size;
  const int B = 4096;
  typedef _Float16 h16;

  const float* x   = (const float*)d_in[0];
  const float* eps = (const float*)d_in[1];
  const float* ew1 = (const float*)d_in[2];
  const float* eb1 = (const float*)d_in[3];
  const float* es1 = (const float*)d_in[4];
  const float* eo1 = (const float*)d_in[5];
  const float* ew2 = (const float*)d_in[6];
  const float* eb2 = (const float*)d_in[7];
  const float* es2 = (const float*)d_in[8];
  const float* eo2 = (const float*)d_in[9];
  const float* ew3 = (const float*)d_in[10];
  const float* eb3 = (const float*)d_in[11];
  const float* es3 = (const float*)d_in[12];
  const float* eo3 = (const float*)d_in[13];
  const float* wzm = (const float*)d_in[14];
  const float* bzm = (const float*)d_in[15];
  const float* wzl = (const float*)d_in[16];
  const float* bzl = (const float*)d_in[17];
  const float* dw  = (const float*)d_in[18];
  const float* db  = (const float*)d_in[19];
  const float* tw1 = (const float*)d_in[20];
  const float* tb1 = (const float*)d_in[21];
  const float* ds1 = (const float*)d_in[22];
  const float* do1 = (const float*)d_in[23];
  const float* tw2 = (const float*)d_in[24];
  const float* tb2 = (const float*)d_in[25];
  const float* ds2 = (const float*)d_in[26];
  const float* do2 = (const float*)d_in[27];
  const float* tw3 = (const float*)d_in[28];
  const float* tb3 = (const float*)d_in[29];
  const float* ds3 = (const float*)d_in[30];
  const float* do3 = (const float*)d_in[31];
  const float* tw4 = (const float*)d_in[32];
  const float* tb4 = (const float*)d_in[33];

  // Two-ended bump arena: producer at one end, consumer at the other.
  char* ws = (char*)d_ws;
  const size_t endA = ws_size & ~(size_t)255;
  auto back = [&](size_t bytes) -> void* {
    return (void*)(ws + ((endA - bytes) & ~(size_t)255));
  };

  const size_t szH2  = (size_t)B * 8 * 8 * 64 * 2;     //  34 MB (f16)
  const size_t szZ   = (size_t)B * 128 * 2;            //   1 MB (f16)
  const size_t szD0  = (size_t)B * 8192 * 2;           //  67 MB (f16)
  const size_t szD2  = (size_t)B * 16 * 16 * 64 * 2;   // 134 MB (f16)
  const size_t szRec = (size_t)B * 32 * 32 * 3 * 2;    //  25 MB (f16)
  const size_t szErr = (size_t)B * 1024 * 4;           //  17 MB (f32)

  h16* h1  = (h16*)ws;             // front
  h16* h2  = (h16*)back(szH2);     // back
  h16* h3  = (h16*)ws;             // front (h1 dead)
  h16* zm  = (h16*)back(szZ);      // back
  h16* zl  = (h16*)back(2 * szZ);  // back (below zm)
  h16* z   = (h16*)ws;             // front (h3 dead)
  h16* d0  = (h16*)back(szD0);     // back
  h16* d1  = (h16*)ws;             // front
  h16* d2  = (h16*)back(szD2);     // back
  h16* d3  = (h16*)ws;             // front (268 MB)
  h16* rec = (h16*)back(szRec);    // back
  float* err = (float*)ws;         // front (d3 dead)
  float* thr = (float*)(ws + szErr);

  // encoder conv1: [B,32,32,3] -> [B,16,16,32], 5x5 s2, SAME(pad_before=1)
  launch_gemm<float, h16, 32, 32, 3, 16, 16, 32, 5, 5, 2, 1, 1, 0, 2>(x, ew1, eb1, es1, eo1, h1, B, stream);
  // encoder conv2: [B,16,16,32] -> [B,8,8,64]
  launch_gemm<h16, h16, 16, 16, 32, 8, 8, 64, 5, 5, 2, 1, 1, 0, 2>(h1, ew2, eb2, es2, eo2, h2, B, stream);
  // encoder conv3: [B,8,8,64] -> [B,4,4,128] (== flat [B,2048])
  launch_gemm<h16, h16, 8, 8, 64, 4, 4, 128, 5, 5, 2, 1, 1, 0, 2>(h2, ew3, eb3, es3, eo3, h3, B, stream);

  // dense heads: z_mean / z_log_var = relu(h @ W + b)  (DENSE fast path)
  launch_gemm<h16, h16, 1, 1, 2048, 1, 1, 128, 1, 1, 1, 0, 1, 0, 1>(h3, wzm, bzm, nullptr, nullptr, zm, B, stream);
  launch_gemm<h16, h16, 1, 1, 2048, 1, 1, 128, 1, 1, 1, 0, 1, 0, 1>(h3, wzl, bzl, nullptr, nullptr, zl, B, stream);

  // sampling
  {
    int n = B * 128;
    hipLaunchKernelGGL(sampling_kernel, dim3((n + 255) / 256), dim3(256), 0, stream,
                       zm, zl, eps, z, n);
  }

  // decoder dense: relu(z @ dw + db) -> [B,8192]  (DENSE fast path)
  launch_gemm<h16, h16, 1, 1, 128, 1, 1, 8192, 1, 1, 1, 0, 1, 0, 1>(z, dw, db, nullptr, nullptr, d0, B, stream);

  // convT1: 1x1 s1 128->128 on [B,8,8,*]  (DENSE fast path)
  launch_gemm<h16, h16, 8, 8, 128, 8, 8, 128, 1, 1, 1, 0, 1, 0, 2>(d0, tw1, tb1, ds1, do1, d1, B, stream);
  // convT2: 3x3 s2 128->64: fractionally-strided conv, ups=2, pad=k-1-0=2, flipped
  launch_gemm<h16, h16, 8, 8, 128, 16, 16, 64, 3, 3, 1, 2, 2, 1, 2>(d1, tw2, tb2, ds2, do2, d2, B, stream);
  // convT3: 3x3 s2 64->32
  launch_gemm<h16, h16, 16, 16, 64, 32, 32, 32, 3, 3, 1, 2, 2, 1, 2>(d2, tw3, tb3, ds3, do3, d3, B, stream);
  // convT4: 3x3 s1 32->3, SAME (pad' = k-1-1 = 1), sigmoid epilogue
  launch_gemm<h16, h16, 32, 32, 32, 32, 32, 3, 3, 3, 1, 1, 1, 1, 3>(d3, tw4, tb4, nullptr, nullptr, rec, B, stream);

  // per-pixel error (f32 x vs f16 recon)
  {
    int n = B * 32 * 32;
    hipLaunchKernelGGL(error_kernel, dim3((n + 255) / 256), dim3(256), 0, stream,
                       x, rec, err, n);
  }
  // per-image Otsu threshold
  hipLaunchKernelGGL(otsu_kernel, dim3(B), dim3(256), 0, stream, err, thr);
  // final int32 mask
  {
    int n = B * 32 * 32;
    hipLaunchKernelGGL(mask_kernel, dim3((n + 255) / 256), dim3(256), 0, stream,
                       err, thr, (int*)d_out, n);
  }
}
